// MultiHeadAttention_29678224015817
// MI455X (gfx1250) — matrix operand order repaired and checked by reading, not compiled, not run on hardware
//
#include <hip/hip_runtime.h>

typedef _Float16 f16t;
typedef _Float16 v16h __attribute__((ext_vector_type(16)));
typedef _Float16 h2v  __attribute__((ext_vector_type(2)));
typedef float    v8f  __attribute__((ext_vector_type(8)));

#define S_LEN   2048
#define D_MODEL 1024
#define NHEAD   16
#define DHEAD   64
#define QTILE   16                  // rows per M-tile
#define MTILES  2                   // M-tiles per wave -> 32 query rows/wave
#define WROWS   (QTILE * MTILES)
#define WAVES   4                   // 128 query rows per block
#define KVT     32                  // kv keys per inner tile
#define KPAD    8
#define PPAD    8
#define VPAD    8

__device__ __forceinline__ h2v cvt2(float x, float y) {
  return __builtin_bit_cast(h2v, __builtin_amdgcn_cvt_pkrtz(x, y));
}

// xor-butterfly within each 16-lane half via v_permlane16_b32
__device__ __forceinline__ float permf(float x, unsigned lo, unsigned hi) {
  int u = __builtin_bit_cast(int, x);
  int r = __builtin_amdgcn_permlane16(u, u, lo, hi, false, false);
  return __builtin_bit_cast(float, r);
}
__device__ __forceinline__ float rowmax16(float x) {
  x = fmaxf(x, permf(x, 0x67452301u, 0xEFCDAB89u));  // xor 1
  x = fmaxf(x, permf(x, 0x45670123u, 0xCDEF89ABu));  // xor 3
  x = fmaxf(x, permf(x, 0x01234567u, 0x89ABCDEFu));  // xor 7
  x = fmaxf(x, permf(x, 0xFEDCBA98u, 0x76543210u));  // xor 8
  return x;
}

// Per-M-tile online-softmax step. qmb is an SGPR-uniform value so the mask
// gate lowers to a scalar branch (no EXEC manipulation around WMMA callers).
__device__ __forceinline__ void softmax_step(
    v8f& s0, v8f& s1, float* mr, v8f* oc,
    f16t (*pl)[KVT + PPAD], int mrow, int half, int kv0, int qmb) {
  if (kv0 + KVT - 1 > qmb) {          // scalar condition: causal tiles only
#pragma unroll
    for (int v = 0; v < 8; ++v) {
      const int qg = qmb + 8 * half + v;
      if (kv0 + mrow      > qg) s0[v] = -1e30f;
      if (kv0 + 16 + mrow > qg) s1[v] = -1e30f;
    }
  }
#pragma unroll
  for (int v = 0; v < 8; ++v) {
    float mx = rowmax16(fmaxf(s0[v], s1[v]));
    const float mnew  = fmaxf(mr[v], mx);
    const float alpha = __expf(mr[v] - mnew);
    mr[v] = mnew;
    const float p0 = __expf(s0[v] - mnew);
    const float p1 = __expf(s1[v] - mnew);
    h2v pp = cvt2(p0, p1);
    pl[8 * half + v][mrow]      = pp[0];
    pl[8 * half + v][16 + mrow] = pp[1];
    oc[0][v] *= alpha; oc[1][v] *= alpha; oc[2][v] *= alpha;
    oc[3][v] *= alpha; oc[4][v] *= alpha;
  }
}

// Causal flash attention: fp32 IO, f16 WMMA operands, f32 accumulation.
// Block = 4 waves x 32 query rows; shared K/V^T staged in LDS once per tile.
__global__ __launch_bounds__(WAVES * 32)
void fa_wmma_kernel(const float* __restrict__ Q, const float* __restrict__ K,
                    const float* __restrict__ V, float* __restrict__ O) {
  __shared__ f16t Klds[KVT][DHEAD + KPAD];                   // K tile, row-major
  __shared__ f16t VTl [DHEAD][KVT + VPAD];                   // V tile, transposed
  __shared__ f16t Plds[WAVES][MTILES][QTILE][KVT + PPAD];    // P staging

  const int tid  = threadIdx.x;
  const int lane = tid & 31;
  const int wave = tid >> 5;
  const int mrow = lane & 15;
  const int half = lane >> 4;

  const int nQt = S_LEN / (WROWS * WAVES);          // 16
  const int qt  = blockIdx.x % nQt;
  const int bh  = blockIdx.x / nQt;
  const int b   = bh / NHEAD;
  const int h   = bh % NHEAD;

  const int qblk  = qt * (WROWS * WAVES);
  // Force wave-uniform query base into an SGPR: causal gates become scalar
  // branches (s_cmp/s_cbranch), keeping EXEC all-ones around every WMMA.
  const int qbase =
      __builtin_amdgcn_readfirstlane(qblk + wave * WROWS);

  const float* Qb = Q + (size_t)b * S_LEN * D_MODEL + h * DHEAD;
  const float* Kb = K + (size_t)b * S_LEN * D_MODEL + h * DHEAD;
  const float* Vb = V + (size_t)b * S_LEN * D_MODEL + h * DHEAD;
  float*       Ob = O + (size_t)b * S_LEN * D_MODEL + h * DHEAD;

  // ---- Q A-fragments, two K=32 chunks per M-tile, pre-scaled by 1/8 ----
  v16h aq[MTILES][2];
#pragma unroll
  for (int mt = 0; mt < MTILES; ++mt) {
    const float* qrow = Qb + (size_t)(qbase + mt * QTILE + mrow) * D_MODEL;
#pragma unroll
    for (int c = 0; c < 2; ++c) {
      const float* p0 = qrow + 32 * c + 16 * half;
      const float* p1 = qrow + 32 * c + 16 * half + 8;
      float4 a  = *(const float4*)(p0);
      float4 bq = *(const float4*)(p0 + 4);
      float4 cq = *(const float4*)(p1);
      float4 d  = *(const float4*)(p1 + 4);
      h2v* ad = (h2v*)&aq[mt][c];
      ad[0] = cvt2(a.x * 0.125f,  a.y * 0.125f);
      ad[1] = cvt2(a.z * 0.125f,  a.w * 0.125f);
      ad[2] = cvt2(bq.x * 0.125f, bq.y * 0.125f);
      ad[3] = cvt2(bq.z * 0.125f, bq.w * 0.125f);
      ad[4] = cvt2(cq.x * 0.125f, cq.y * 0.125f);
      ad[5] = cvt2(cq.z * 0.125f, cq.w * 0.125f);
      ad[6] = cvt2(d.x * 0.125f,  d.y * 0.125f);
      ad[7] = cvt2(d.z * 0.125f,  d.w * 0.125f);
    }
  }

  // Ones B-fragment (column 0 all ones) -> softmax denominator via WMMA.
  v16h bones;
#pragma unroll
  for (int e = 0; e < 16; ++e) bones[e] = (mrow == 0) ? (f16t)1.0f : (f16t)0.0f;

  const v8f zacc = {0,0,0,0,0,0,0,0};
  v8f oacc[MTILES][5];                 // [..][4] = row-sum accumulator
#pragma unroll
  for (int mt = 0; mt < MTILES; ++mt)
#pragma unroll
    for (int t = 0; t < 5; ++t) oacc[mt][t] = zacc;
  float mrun[MTILES][8];
#pragma unroll
  for (int mt = 0; mt < MTILES; ++mt)
#pragma unroll
    for (int v = 0; v < 8; ++v) mrun[mt][v] = -1e30f;

  // staging assignments (128 threads, 32x64 tile)
  const int krow_s = tid >> 2;            // 0..31
  const int kcol_s = (tid & 3) * 16;      // 0,16,32,48
  const int vrow_s = tid & 31;            // 0..31
  const int vcol_s = (tid >> 5) * 16;     // 0,16,32,48

  const int jmax = (qblk + WROWS * WAVES - 1) / KVT;   // block causal bound
  for (int j = 0; j <= jmax; ++j) {
    const int kv0 = j * KVT;
    const float* kg = Kb + (size_t)(kv0 + krow_s) * D_MODEL + kcol_s;
    const float* vg = Vb + (size_t)(kv0 + vrow_s) * D_MODEL + vcol_s;

    if (j < jmax) {
      __builtin_prefetch(kg + (size_t)KVT * D_MODEL, 0, 3);
      __builtin_prefetch(vg + (size_t)KVT * D_MODEL, 0, 3);
    }

    // ---- Cooperative staging: K (row-major f16) and V^T ----
    {
      float4 k0 = *(const float4*)(kg + 0);
      float4 k1 = *(const float4*)(kg + 4);
      float4 k2 = *(const float4*)(kg + 8);
      float4 k3 = *(const float4*)(kg + 12);
      h2v* kd = (h2v*)&Klds[krow_s][kcol_s];
      kd[0] = cvt2(k0.x, k0.y); kd[1] = cvt2(k0.z, k0.w);
      kd[2] = cvt2(k1.x, k1.y); kd[3] = cvt2(k1.z, k1.w);
      kd[4] = cvt2(k2.x, k2.y); kd[5] = cvt2(k2.z, k2.w);
      kd[6] = cvt2(k3.x, k3.y); kd[7] = cvt2(k3.z, k3.w);

      float4 v0 = *(const float4*)(vg + 0);
      float4 v1 = *(const float4*)(vg + 4);
      float4 v2 = *(const float4*)(vg + 8);
      float4 v3 = *(const float4*)(vg + 12);
      VTl[vcol_s +  0][vrow_s] = (f16t)v0.x; VTl[vcol_s +  1][vrow_s] = (f16t)v0.y;
      VTl[vcol_s +  2][vrow_s] = (f16t)v0.z; VTl[vcol_s +  3][vrow_s] = (f16t)v0.w;
      VTl[vcol_s +  4][vrow_s] = (f16t)v1.x; VTl[vcol_s +  5][vrow_s] = (f16t)v1.y;
      VTl[vcol_s +  6][vrow_s] = (f16t)v1.z; VTl[vcol_s +  7][vrow_s] = (f16t)v1.w;
      VTl[vcol_s +  8][vrow_s] = (f16t)v2.x; VTl[vcol_s +  9][vrow_s] = (f16t)v2.y;
      VTl[vcol_s + 10][vrow_s] = (f16t)v2.z; VTl[vcol_s + 11][vrow_s] = (f16t)v2.w;
      VTl[vcol_s + 12][vrow_s] = (f16t)v3.x; VTl[vcol_s + 13][vrow_s] = (f16t)v3.y;
      VTl[vcol_s + 14][vrow_s] = (f16t)v3.z; VTl[vcol_s + 15][vrow_s] = (f16t)v3.w;
    }
    __syncthreads();

    // Scalar (SGPR) causal gates -> s_cbranch, EXEC untouched around WMMA.
    const bool act0 = kv0 <= qbase + QTILE - 1;
    const bool act1 = kv0 <= qbase + WROWS - 1;
    if (act1) {
      // ---- S = Q*K^T : K B-frags shared across both M-tiles ----
      v8f s[MTILES][2];
#pragma unroll
      for (int nh = 0; nh < 2; ++nh) {
        v16h bk0, bk1;
        const f16t* kp = &Klds[16 * nh + mrow][16 * half];
#pragma unroll
        for (int e = 0; e < 16; ++e) bk0[e] = kp[e];
#pragma unroll
        for (int e = 0; e < 16; ++e) bk1[e] = kp[32 + e];
        v8f a1 = __builtin_amdgcn_wmma_f32_16x16x32_f16(
            false, aq[1][0], false, bk0, (short)0, zacc, false, false);
        s[1][nh] = __builtin_amdgcn_wmma_f32_16x16x32_f16(
            false, aq[1][1], false, bk1, (short)0, a1, false, false);
        if (act0) {
          v8f a0 = __builtin_amdgcn_wmma_f32_16x16x32_f16(
              false, aq[0][0], false, bk0, (short)0, zacc, false, false);
          s[0][nh] = __builtin_amdgcn_wmma_f32_16x16x32_f16(
              false, aq[0][1], false, bk1, (short)0, a0, false, false);
        }
      }

      // ---- Online softmax per M-tile ----
      softmax_step(s[1][0], s[1][1], mrun[1], oacc[1],
                   Plds[wave][1], mrow, half, kv0, qbase + QTILE);
      if (act0)
        softmax_step(s[0][0], s[0][1], mrun[0], oacc[0],
                     Plds[wave][0], mrow, half, kv0, qbase);

      // Drain per-wave DS writes (Plds cross-lane handoff).
      asm volatile("s_wait_dscnt 0" ::: "memory");

      // ---- P A-fragments ----
      v16h ap[MTILES];
#pragma unroll
      for (int mt = 0; mt < MTILES; ++mt) {
        const f16t* prow = &Plds[wave][mt][mrow][0];
#pragma unroll
        for (int e = 0; e < 8; ++e) ap[mt][e]     = prow[16 * half + e];
#pragma unroll
        for (int e = 0; e < 8; ++e) ap[mt][8 + e] = prow[16 * half + 8 + e];
      }

      // ---- O += P*V (V B-frags shared across M-tiles); row-sum via ones ----
#pragma unroll
      for (int t = 0; t < 4; ++t) {
        v16h bv;
        const f16t* vtrow = &VTl[16 * t + mrow][16 * half];
#pragma unroll
        for (int e = 0; e < 16; ++e) bv[e] = vtrow[e];
        oacc[1][t] = __builtin_amdgcn_wmma_f32_16x16x32_f16(
            false, ap[1], false, bv, (short)0, oacc[1][t], false, false);
        if (act0)
          oacc[0][t] = __builtin_amdgcn_wmma_f32_16x16x32_f16(
              false, ap[0], false, bv, (short)0, oacc[0][t], false, false);
      }
      oacc[1][4] = __builtin_amdgcn_wmma_f32_16x16x32_f16(
          false, ap[1], false, bones, (short)0, oacc[1][4], false, false);
      if (act0)
        oacc[0][4] = __builtin_amdgcn_wmma_f32_16x16x32_f16(
            false, ap[0], false, bones, (short)0, oacc[0][4], false, false);
    }
    __syncthreads();   // protect Klds/VTl before next tile's staging
  }

  // ---- Epilogue: O /= rowsum (broadcast from column-0 lanes of oacc[..][4])
#pragma unroll
  for (int mt = 0; mt < MTILES; ++mt) {
#pragma unroll
    for (int v = 0; v < 8; ++v) {
      const float lsum = __shfl(oacc[mt][4][v], 16 * half, 32);
      const float inv  = 1.0f / lsum;
      float* orow =
          Ob + (size_t)(qbase + mt * QTILE + 8 * half + v) * D_MODEL;
      orow[mrow]      = oacc[mt][0][v] * inv;
      orow[16 + mrow] = oacc[mt][1][v] * inv;
      orow[32 + mrow] = oacc[mt][2][v] * inv;
      orow[48 + mrow] = oacc[mt][3][v] * inv;
    }
  }
}

extern "C" void kernel_launch(void* const* d_in, const int* in_sizes, int n_in,
                              void* d_out, int out_size, void* d_ws, size_t ws_size,
                              hipStream_t stream) {
  const float* q = (const float*)d_in[0];
  const float* k = (const float*)d_in[1];
  const float* v = (const float*)d_in[2];
  // d_in[3] (mask) / d_in[4] (training): causal mask is implemented implicitly.
  float* out = (float*)d_out;

  const int B = 4, H = 16;
  dim3 grid(B * H * (S_LEN / (WROWS * WAVES)));   // 1024 blocks
  dim3 block(WAVES * 32);                          // 4 waves of 32
  fa_wmma_kernel<<<grid, block, 0, stream>>>(q, k, v, out);
}
